// MultiHeadSNNTransformerBlock_77223511982132
// MI455X (gfx1250) — matrix operand order repaired and checked by reading, not compiled, not run on hardware
//
#include <hip/hip_runtime.h>
#include <hip/hip_bf16.h>
#include <cstdint>
#include <cstddef>

#define BB 4
#define NN 4096
#define KK 16
#define CM 128
#define CP 64

typedef __attribute__((ext_vector_type(16))) __bf16 bf16x16;
typedef __attribute__((ext_vector_type(8)))  float  f32x8;

union FragB { bf16x16 v; uint4 q[2]; };

__device__ __forceinline__ unsigned short f2bf(float f) {
    unsigned u = __float_as_uint(f);
    u += 0x7fffu + ((u >> 16) & 1u);
    return (unsigned short)(u >> 16);
}
__device__ __forceinline__ float bfl(unsigned u) { return __uint_as_float(u << 16); }
__device__ __forceinline__ float bfh(unsigned u) { return __uint_as_float(u & 0xffff0000u); }
__device__ __forceinline__ unsigned pack2(float lo, float hi) {
    return (unsigned)f2bf(lo) | ((unsigned)f2bf(hi) << 16);
}
__device__ __forceinline__ uint4 pack8(const float* v) {
    return make_uint4(pack2(v[0], v[1]), pack2(v[2], v[3]), pack2(v[4], v[5]), pack2(v[6], v[7]));
}
__device__ __forceinline__ void unpack8(uint4 u, float* v) {
    v[0] = bfl(u.x); v[1] = bfh(u.x); v[2] = bfl(u.y); v[3] = bfh(u.y);
    v[4] = bfl(u.z); v[5] = bfh(u.z); v[6] = bfl(u.w); v[7] = bfh(u.w);
}

__device__ __forceinline__ f32x8 wmma_bf16(const FragB& a, const FragB& b, f32x8 c) {
    return __builtin_amdgcn_wmma_f32_16x16x32_bf16(false, a.v, false, b.v, (short)0, c, false, false);
}

// Eval-mode LIF sequence, T=4, spikes fed back as input.
__device__ __forceinline__ float lif_apply(float x, float md, float ta, float rd, float tb) {
    float mem = 0.f, thr = tb, ref = 0.f;
#pragma unroll
    for (int t = 0; t < 4; ++t) {
        float xi = x * ((ref <= 0.f) ? 1.f : 0.f);
        mem = mem * md * (1.f - ref) + xi;
        float u = mem - thr;
        u = fminf(fmaxf(u, -10.f), 10.f);
        float s = 0.5f * 0.3989422804014327f * __expf(-0.5f * u * u)
                + 0.5f / (1.f + __expf(-10.f * u));
        mem = mem * (1.f - s);
        ref = ref * rd + s;
        thr = tb + (thr + ta * s - tb) * 0.95f;
        x = s;
    }
    return x;
}

// ---------------- param prep ----------------
__global__ void cvt_bf16_kernel(const float* __restrict__ s, __bf16* __restrict__ d, int n) {
    int i = blockIdx.x * blockDim.x + threadIdx.x;
    if (i < n) ((unsigned short*)d)[i] = f2bf(s[i]);
}

__global__ void prep_bn_kernel(const float* b, const float* g, const float* beta,
                               const float* m, const float* v, float* scale, float* shift, int c) {
    int t = threadIdx.x;
    if (t < c) {
        float s = g[t] * rsqrtf(v[t] + 1e-5f);
        scale[t] = s;
        shift[t] = (b[t] - m[t]) * s + beta[t];
    }
}

__global__ void prep_lif_kernel(const float* md, const float* ta, const float* rd,
                                const float* tb, float* dst) {
    int t = threadIdx.x;
    if (t < 128) {
        dst[t]       = fminf(fmaxf(md[t], 0.1f), 0.99f);
        dst[128 + t] = fminf(fmaxf(ta[t], 0.001f), 0.1f);
        dst[256 + t] = fminf(fmaxf(rd[t], 0.1f), 0.95f);
        dst[384 + t] = tb[t];
    }
}

// ---------------- KNN (brute force, top-16 in registers) ----------------
__global__ __launch_bounds__(128) void knn_kernel(const float* __restrict__ xyz, int* __restrict__ idx) {
    __shared__ float sx[128], sy[128], sz[128];
    int b  = blockIdx.x / (NN / 128);
    int nq = (blockIdx.x % (NN / 128)) * 128 + threadIdx.x;
    const float* base = xyz + (size_t)b * NN * 3;
    float qx = base[nq * 3], qy = base[nq * 3 + 1], qz = base[nq * 3 + 2];
    float bd[16];
    int   bi[16];
#pragma unroll
    for (int i = 0; i < 16; ++i) { bd[i] = 3.4e38f; bi[i] = 0; }
    float worst = 3.4e38f;
    int   wslot = 0;
    for (int tile = 0; tile < NN; tile += 128) {
        __syncthreads();
        int j = tile + threadIdx.x;
        sx[threadIdx.x] = base[j * 3];
        sy[threadIdx.x] = base[j * 3 + 1];
        sz[threadIdx.x] = base[j * 3 + 2];
        __syncthreads();
        for (int t = 0; t < 128; ++t) {
            float dx = qx - sx[t], dy = qy - sy[t], dz = qz - sz[t];
            float d = dx * dx + dy * dy + dz * dz;
            if (d < worst) {
#pragma unroll
                for (int i = 0; i < 16; ++i)
                    if (i == wslot) { bd[i] = d; bi[i] = tile + t; }
                worst = bd[0]; wslot = 0;
#pragma unroll
                for (int i = 1; i < 16; ++i)
                    if (bd[i] > worst) { worst = bd[i]; wslot = i; }
            }
        }
    }
    int* o = idx + ((size_t)b * NN + nq) * KK;
#pragma unroll
    for (int i = 0; i < 16; ++i) o[i] = bi[i];
}

// ---------------- generic WMMA GEMM + BN (+LIF), bf16 in/out ----------------
// X: [P, CIN] bf16 (channel-contiguous), Wg: [COUT, CIN] bf16, Y: [P, COUT] bf16.
// One wave = 16 positions x COUT channels. Block = 8 waves = 128 positions.
template <int CIN, int COUT, bool DO_LIF>
__global__ __launch_bounds__(256) void gemm_bn_lif_kernel(
    const __bf16* __restrict__ X, const __bf16* __restrict__ Wg,
    const float* __restrict__ sc, const float* __restrict__ sh,
    const float* __restrict__ lp, __bf16* __restrict__ Y) {
    __shared__ __bf16 sW[COUT * CIN];
    int tid = threadIdx.x;
    {
        const uint4* s = (const uint4*)Wg;
        uint4*       d = (uint4*)sW;
        for (int i = tid; i < COUT * CIN / 8; i += 256) d[i] = s[i];
    }
    __syncthreads();

    int lane = tid & 31;
    int row = lane & 15, half = lane >> 4;
    size_t p = (size_t)(((size_t)blockIdx.x * 256 + tid) >> 5) * 16 + row;

    constexpr int MT = COUT / 16;
    f32x8 acc[MT];
#pragma unroll
    for (int m = 0; m < MT; ++m)
#pragma unroll
        for (int r = 0; r < 8; ++r) acc[m][r] = 0.f;

    __builtin_prefetch(X + p * CIN, 0, 3);
#pragma unroll
    for (int kk = 0; kk < CIN / 32; ++kk) {
        FragB bfr;
        const uint4* xb = (const uint4*)(X + p * CIN + kk * 32 + half * 16);
        bfr.q[0] = xb[0];
        bfr.q[1] = xb[1];
#pragma unroll
        for (int m = 0; m < MT; ++m) {
            FragB afr;
            const uint4* wb = (const uint4*)(sW + (m * 16 + row) * CIN + kk * 32 + half * 16);
            afr.q[0] = wb[0];
            afr.q[1] = wb[1];
            acc[m] = wmma_bf16(afr, bfr, acc[m]);
        }
    }
#pragma unroll
    for (int m = 0; m < MT; ++m) {
        int co0 = m * 16 + half * 8;
        float v[8];
#pragma unroll
        for (int r = 0; r < 8; ++r) v[r] = acc[m][r] * sc[co0 + r] + sh[co0 + r];
        if constexpr (DO_LIF) {
#pragma unroll
            for (int r = 0; r < 8; ++r)
                v[r] = lif_apply(v[r], lp[co0 + r], lp[128 + co0 + r], lp[256 + co0 + r], lp[384 + co0 + r]);
        }
        *(uint4*)(Y + p * COUT + co0) = pack8(v);
    }
}

// ---------------- fused pos_diff -> fc_delta+LIF -> fc_delta2+LIF ----------------
// Block handles 128 (n,k) columns. Phase1: scalar 3->128 conv + LIF into LDS tile.
// Phase2: WMMA 128->128 from LDS tile, BN+LIF, store pe bf16.
__global__ __launch_bounds__(256) void pe_kernel(
    const float* __restrict__ xyz, const int* __restrict__ idxB,
    const float* __restrict__ Wd, const __bf16* __restrict__ W2,
    const float* __restrict__ scD, const float* __restrict__ shD, const float* __restrict__ lifD,
    const float* __restrict__ sc2, const float* __restrict__ sh2, const float* __restrict__ lif2,
    __bf16* __restrict__ peOut) {
    __shared__ __bf16 sP[128 * CM];
    int tid = threadIdx.x;
    {
        int lcol = tid & 127, cbase = (tid >> 7) * 64;
        size_t col = (size_t)blockIdx.x * 128 + lcol;
        int np = (int)(col >> 4);
        int b = np >> 12, nq = np & (NN - 1);
        int nb = idxB[col];
        const float* xb = xyz + (size_t)b * NN * 3;
        float pdx = xb[nq * 3] - xb[nb * 3];
        float pdy = xb[nq * 3 + 1] - xb[nb * 3 + 1];
        float pdz = xb[nq * 3 + 2] - xb[nb * 3 + 2];
        for (int c = cbase; c < cbase + 64; ++c) {
            float val = Wd[c * 3] * pdx + Wd[c * 3 + 1] * pdy + Wd[c * 3 + 2] * pdz;
            val = val * scD[c] + shD[c];
            val = lif_apply(val, lifD[c], lifD[128 + c], lifD[256 + c], lifD[384 + c]);
            ((unsigned short*)sP)[lcol * CM + c] = f2bf(val);
        }
    }
    __syncthreads();

    int lane = tid & 31, wid = tid >> 5;
    int row = lane & 15, half = lane >> 4;
    f32x8 acc[8];
#pragma unroll
    for (int m = 0; m < 8; ++m)
#pragma unroll
        for (int r = 0; r < 8; ++r) acc[m][r] = 0.f;

    const __bf16* bcol = sP + (wid * 16 + row) * CM;
#pragma unroll
    for (int kk = 0; kk < 4; ++kk) {
        FragB bfr;
        const uint4* xb = (const uint4*)(bcol + kk * 32 + half * 16);
        bfr.q[0] = xb[0];
        bfr.q[1] = xb[1];
#pragma unroll
        for (int m = 0; m < 8; ++m) {
            FragB afr;
            const uint4* wb = (const uint4*)(W2 + (m * 16 + row) * CM + kk * 32 + half * 16);
            afr.q[0] = wb[0];
            afr.q[1] = wb[1];
            acc[m] = wmma_bf16(afr, bfr, acc[m]);
        }
    }
    size_t gcol = (size_t)blockIdx.x * 128 + wid * 16 + row;
#pragma unroll
    for (int m = 0; m < 8; ++m) {
        int co0 = m * 16 + half * 8;
        float v[8];
#pragma unroll
        for (int r = 0; r < 8; ++r) {
            v[r] = acc[m][r] * sc2[co0 + r] + sh2[co0 + r];
            v[r] = lif_apply(v[r], lif2[co0 + r], lif2[128 + co0 + r], lif2[256 + co0 + r], lif2[384 + co0 + r]);
        }
        *(uint4*)(peOut + gcol * CM + co0) = pack8(v);
    }
}

__device__ __forceinline__ unsigned comb_qkp(unsigned q, unsigned k, unsigned p) {
    float lo = bfl(q) - bfl(k) + bfl(p);
    float hi = bfh(q) - bfh(k) + bfh(p);
    return pack2(lo, hi);
}
__device__ __forceinline__ uint4 comb4(uint4 q, uint4 k, uint4 p) {
    return make_uint4(comb_qkp(q.x, k.x, p.x), comb_qkp(q.y, k.y, p.y),
                      comb_qkp(q.z, k.z, p.z), comb_qkp(q.w, k.w, p.w));
}

// ---------------- fc_gamma with gather-fused B fragments: attn_in = q - k_nb + pe ----
__global__ __launch_bounds__(256) void gamma_kernel(
    const __bf16* __restrict__ qb, const __bf16* __restrict__ kb, const __bf16* __restrict__ peb,
    const int* __restrict__ idxB, const __bf16* __restrict__ Wg,
    const float* __restrict__ sc, const float* __restrict__ sh, const float* __restrict__ lp,
    __bf16* __restrict__ gOut) {
    __shared__ __bf16 sW[CM * CM];
    int tid = threadIdx.x;
    {
        const uint4* s = (const uint4*)Wg;
        uint4*       d = (uint4*)sW;
        for (int i = tid; i < CM * CM / 8; i += 256) d[i] = s[i];
    }
    __syncthreads();

    int lane = tid & 31;
    int row = lane & 15, half = lane >> 4;
    size_t col = (size_t)(((size_t)blockIdx.x * 256 + tid) >> 5) * 16 + row;
    int np = (int)(col >> 4);
    int b = np >> 12;
    size_t nbp = (size_t)b * NN + idxB[col];
    const __bf16* qrow = qb + (size_t)np * CM;
    const __bf16* krow = kb + nbp * CM;
    const __bf16* prow = peb + col * CM;

    f32x8 acc[8];
#pragma unroll
    for (int m = 0; m < 8; ++m)
#pragma unroll
        for (int r = 0; r < 8; ++r) acc[m][r] = 0.f;

#pragma unroll
    for (int kk = 0; kk < 4; ++kk) {
        int kb0 = kk * 32 + half * 16;
        uint4 q0 = *(const uint4*)(qrow + kb0), q1 = *(const uint4*)(qrow + kb0 + 8);
        uint4 n0 = *(const uint4*)(krow + kb0), n1 = *(const uint4*)(krow + kb0 + 8);
        uint4 p0 = *(const uint4*)(prow + kb0), p1 = *(const uint4*)(prow + kb0 + 8);
        FragB bfr;
        bfr.q[0] = comb4(q0, n0, p0);
        bfr.q[1] = comb4(q1, n1, p1);
#pragma unroll
        for (int m = 0; m < 8; ++m) {
            FragB afr;
            const uint4* wb = (const uint4*)(sW + (m * 16 + row) * CM + kk * 32 + half * 16);
            afr.q[0] = wb[0];
            afr.q[1] = wb[1];
            acc[m] = wmma_bf16(afr, bfr, acc[m]);
        }
    }
#pragma unroll
    for (int m = 0; m < 8; ++m) {
        int co0 = m * 16 + half * 8;
        float v[8];
#pragma unroll
        for (int r = 0; r < 8; ++r) {
            v[r] = acc[m][r] * sc[co0 + r] + sh[co0 + r];
            v[r] = lif_apply(v[r], lp[co0 + r], lp[128 + co0 + r], lp[256 + co0 + r], lp[384 + co0 + r]);
        }
        *(uint4*)(gOut + col * CM + co0) = pack8(v);
    }
}

// ---------------- fc_gamma2 + softmax over K + weighted neighbor sum ----------------
// One wave = one point n: its K=16 neighbors are exactly one 16-column WMMA tile.
__global__ __launch_bounds__(256) void gamma2_kernel(
    const __bf16* __restrict__ gb, const __bf16* __restrict__ vb, const __bf16* __restrict__ peb,
    const int* __restrict__ idxB, const __bf16* __restrict__ Wg2,
    const float* __restrict__ sc, const float* __restrict__ sh,
    __bf16* __restrict__ resOut) {
    __shared__ __bf16 sW[CM * CM];
    int tid = threadIdx.x;
    {
        const uint4* s = (const uint4*)Wg2;
        uint4*       d = (uint4*)sW;
        for (int i = tid; i < CM * CM / 8; i += 256) d[i] = s[i];
    }
    __syncthreads();

    int lane = tid & 31, wid = tid >> 5;
    int row = lane & 15, half = lane >> 4;       // row == neighbor index k
    size_t np = (size_t)blockIdx.x * 8 + wid;    // point index
    size_t col = np * 16 + row;
    int b = (int)(np >> 12);
    size_t nbp = (size_t)b * NN + idxB[col];
    const __bf16* grow = gb + col * CM;

    f32x8 acc[8];
#pragma unroll
    for (int m = 0; m < 8; ++m)
#pragma unroll
        for (int r = 0; r < 8; ++r) acc[m][r] = 0.f;

#pragma unroll
    for (int kk = 0; kk < 4; ++kk) {
        FragB bfr;
        const uint4* xb = (const uint4*)(grow + kk * 32 + half * 16);
        bfr.q[0] = xb[0];
        bfr.q[1] = xb[1];
#pragma unroll
        for (int m = 0; m < 8; ++m) {
            FragB afr;
            const uint4* wb = (const uint4*)(sW + (m * 16 + row) * CM + kk * 32 + half * 16);
            afr.q[0] = wb[0];
            afr.q[1] = wb[1];
            acc[m] = wmma_bf16(afr, bfr, acc[m]);
        }
    }

    const float invs = 0.17677669529663687f;  // 1/sqrt(HD=32)
#pragma unroll
    for (int m = 0; m < 8; ++m) {
        int co0 = m * 16 + half * 8;
        float t[8], at[8];
#pragma unroll
        for (int r = 0; r < 8; ++r) t[r] = (acc[m][r] * sc[co0 + r] + sh[co0 + r]) * invs;
#pragma unroll
        for (int r = 0; r < 8; ++r) {
            float mx = t[r];
#pragma unroll
            for (int s = 1; s < 16; s <<= 1) mx = fmaxf(mx, __shfl_xor(mx, s, 32));
            float e = __expf(t[r] - mx);
            float sm = e;
#pragma unroll
            for (int s = 1; s < 16; s <<= 1) sm += __shfl_xor(sm, s, 32);
            at[r] = e / sm;
        }
        float vh[8], ph[8];
        unpack8(*(const uint4*)(vb + nbp * CM + co0), vh);
        unpack8(*(const uint4*)(peb + col * CM + co0), ph);
        float tot[8];
#pragma unroll
        for (int r = 0; r < 8; ++r) {
            float sv = at[r] * (vh[r] + ph[r]);
#pragma unroll
            for (int s = 1; s < 16; s <<= 1) sv += __shfl_xor(sv, s, 32);
            tot[r] = sv;
        }
        if (row == 0) *(uint4*)(resOut + np * CM + co0) = pack8(tot);
    }
}

// ---------------- fc2 + BN + residual -> f32 output [B,N,64] ----------------
__global__ __launch_bounds__(256) void fc2_final_kernel(
    const __bf16* __restrict__ X, const __bf16* __restrict__ Wg,
    const float* __restrict__ sc, const float* __restrict__ sh,
    const float* __restrict__ feats, float* __restrict__ out) {
    __shared__ __bf16 sW[CP * CM];
    int tid = threadIdx.x;
    {
        const uint4* s = (const uint4*)Wg;
        uint4*       d = (uint4*)sW;
        for (int i = tid; i < CP * CM / 8; i += 256) d[i] = s[i];
    }
    __syncthreads();

    int lane = tid & 31;
    int row = lane & 15, half = lane >> 4;
    size_t p = (size_t)(((size_t)blockIdx.x * 256 + tid) >> 5) * 16 + row;

    f32x8 acc[4];
#pragma unroll
    for (int m = 0; m < 4; ++m)
#pragma unroll
        for (int r = 0; r < 8; ++r) acc[m][r] = 0.f;

#pragma unroll
    for (int kk = 0; kk < 4; ++kk) {
        FragB bfr;
        const uint4* xb = (const uint4*)(X + p * CM + kk * 32 + half * 16);
        bfr.q[0] = xb[0];
        bfr.q[1] = xb[1];
#pragma unroll
        for (int m = 0; m < 4; ++m) {
            FragB afr;
            const uint4* wb = (const uint4*)(sW + (m * 16 + row) * CM + kk * 32 + half * 16);
            afr.q[0] = wb[0];
            afr.q[1] = wb[1];
            acc[m] = wmma_bf16(afr, bfr, acc[m]);
        }
    }
#pragma unroll
    for (int m = 0; m < 4; ++m) {
        int co0 = m * 16 + half * 8;
        float4 f0 = *(const float4*)(feats + p * CP + co0);
        float4 f1 = *(const float4*)(feats + p * CP + co0 + 4);
        float v[8];
#pragma unroll
        for (int r = 0; r < 8; ++r) v[r] = acc[m][r] * sc[co0 + r] + sh[co0 + r];
        v[0] += f0.x; v[1] += f0.y; v[2] += f0.z; v[3] += f0.w;
        v[4] += f1.x; v[5] += f1.y; v[6] += f1.z; v[7] += f1.w;
        *(float4*)(out + p * CP + co0)     = make_float4(v[0], v[1], v[2], v[3]);
        *(float4*)(out + p * CP + co0 + 4) = make_float4(v[4], v[5], v[6], v[7]);
    }
}

// ---------------------------------------------------------------------------
extern "C" void kernel_launch(void* const* d_in, const int* in_sizes, int n_in,
                              void* d_out, int out_size, void* d_ws, size_t ws_size,
                              hipStream_t stream) {
    (void)in_sizes; (void)n_in; (void)out_size; (void)ws_size;
    // d_in flattening (insertion order): xyz, features, then params:
    // convs {W,b,g,beta,m,v}: fc1@2 fc2@8 fc_delta@14 fc_delta2@20 fc_gamma@26
    //   fc_gamma2@32 w_qs@38 w_ks@44 w_vs@50 out_proj@56
    // lifs {md,ta,rd,tb}: snn1@62 snn_q@66 snn_k@70 snn_v@74 snn_delta@78 snn_delta2@82 snn_gamma@86
    const float* xyz   = (const float*)d_in[0];
    const float* feats = (const float*)d_in[1];
    auto F = [&](int i) { return (const float*)d_in[i]; };

    char* wsb = (char*)d_ws;
    size_t off = 0;
    auto alloc = [&](size_t bytes) -> void* {
        void* p = wsb + off;
        off += (bytes + 255) & ~(size_t)255;
        return p;
    };
    const size_t PN = (size_t)BB * NN;           // 16384 points
    const size_t PC = PN * KK;                   // 262144 (n,k) columns

    int*    idxB   = (int*)alloc(PC * 4);
    __bf16* featbf = (__bf16*)alloc(PN * CP * 2);
    __bf16* xbf    = (__bf16*)alloc(PN * CM * 2);
    __bf16* qbf    = (__bf16*)alloc(PN * CM * 2);
    __bf16* kbf    = (__bf16*)alloc(PN * CM * 2);
    __bf16* vbf    = (__bf16*)alloc(PN * CM * 2);
    __bf16* pebf   = (__bf16*)alloc(PC * CM * 2);
    __bf16* gbf    = (__bf16*)alloc(PC * CM * 2);
    __bf16* resbf  = (__bf16*)alloc(PN * CM * 2);
    __bf16* midbf  = (__bf16*)alloc(PN * CM * 2);
    __bf16* wFc1 = (__bf16*)alloc(CM * CP * 2);
    __bf16* wQs  = (__bf16*)alloc(CM * CM * 2);
    __bf16* wKs  = (__bf16*)alloc(CM * CM * 2);
    __bf16* wVs  = (__bf16*)alloc(CM * CM * 2);
    __bf16* wD2  = (__bf16*)alloc(CM * CM * 2);
    __bf16* wG1  = (__bf16*)alloc(CM * CM * 2);
    __bf16* wG2  = (__bf16*)alloc(CM * CM * 2);
    __bf16* wOp  = (__bf16*)alloc(CM * CM * 2);
    __bf16* wFc2 = (__bf16*)alloc(CP * CM * 2);
    float* bnTab  = (float*)alloc(10 * 256 * 4);  // per conv: scale[128] | shift[128]
    float* lifTab = (float*)alloc(7 * 512 * 4);   // per lif: md|ta|rd|tb x 128

    // ---- param prep ----
    const int convBase[10] = {2, 8, 14, 20, 26, 32, 38, 44, 50, 56};
    const int convCout[10] = {128, 64, 128, 128, 128, 128, 128, 128, 128, 128};
    for (int c = 0; c < 10; ++c) {
        int bs = convBase[c];
        prep_bn_kernel<<<1, 128, 0, stream>>>(F(bs + 1), F(bs + 2), F(bs + 3), F(bs + 4), F(bs + 5),
                                              bnTab + c * 256, bnTab + c * 256 + 128, convCout[c]);
    }
    const int lifBase[7] = {62, 66, 70, 74, 78, 82, 86};
    for (int l = 0; l < 7; ++l) {
        int bs = lifBase[l];
        prep_lif_kernel<<<1, 128, 0, stream>>>(F(bs), F(bs + 1), F(bs + 2), F(bs + 3), lifTab + l * 512);
    }
    auto cvt = [&](const float* src, __bf16* dst, int n) {
        cvt_bf16_kernel<<<(n + 255) / 256, 256, 0, stream>>>(src, dst, n);
    };
    cvt(F(2),  wFc1, CM * CP);
    cvt(F(38), wQs,  CM * CM);
    cvt(F(44), wKs,  CM * CM);
    cvt(F(50), wVs,  CM * CM);
    cvt(F(20), wD2,  CM * CM);
    cvt(F(26), wG1,  CM * CM);
    cvt(F(32), wG2,  CM * CM);
    cvt(F(56), wOp,  CM * CM);
    cvt(F(8),  wFc2, CP * CM);
    cvt(feats, featbf, (int)(PN * CP));

    // ---- KNN ----
    knn_kernel<<<BB * (NN / 128), 128, 0, stream>>>(xyz, idxB);

    // ---- fc1 + LIF -> x ----
    gemm_bn_lif_kernel<CP, CM, true><<<(int)(PN / 128), 256, 0, stream>>>(
        featbf, wFc1, bnTab + 0 * 256, bnTab + 0 * 256 + 128, lifTab + 0 * 512, xbf);
    // ---- q/k/v ----
    gemm_bn_lif_kernel<CM, CM, true><<<(int)(PN / 128), 256, 0, stream>>>(
        xbf, wQs, bnTab + 6 * 256, bnTab + 6 * 256 + 128, lifTab + 1 * 512, qbf);
    gemm_bn_lif_kernel<CM, CM, true><<<(int)(PN / 128), 256, 0, stream>>>(
        xbf, wKs, bnTab + 7 * 256, bnTab + 7 * 256 + 128, lifTab + 2 * 512, kbf);
    gemm_bn_lif_kernel<CM, CM, true><<<(int)(PN / 128), 256, 0, stream>>>(
        xbf, wVs, bnTab + 8 * 256, bnTab + 8 * 256 + 128, lifTab + 3 * 512, vbf);

    // ---- pe = lif(fc_delta2(lif(fc_delta(pos_diff)))) ----
    pe_kernel<<<(int)(PC / 128), 256, 0, stream>>>(
        xyz, idxB, F(14), wD2,
        bnTab + 2 * 256, bnTab + 2 * 256 + 128, lifTab + 4 * 512,
        bnTab + 3 * 256, bnTab + 3 * 256 + 128, lifTab + 5 * 512, pebf);

    // ---- g = lif(fc_gamma(q - k_nb + pe)) (gather fused) ----
    gamma_kernel<<<(int)(PC / 128), 256, 0, stream>>>(
        qbf, kbf, pebf, idxB, wG1,
        bnTab + 4 * 256, bnTab + 4 * 256 + 128, lifTab + 6 * 512, gbf);

    // ---- fc_gamma2 + softmax over K + weighted sum of (v_nb + pe) ----
    gamma2_kernel<<<(int)(PN / 8), 256, 0, stream>>>(
        gbf, vbf, pebf, idxB, wG2, bnTab + 5 * 256, bnTab + 5 * 256 + 128, resbf);

    // ---- out_proj (no LIF) ----
    gemm_bn_lif_kernel<CM, CM, false><<<(int)(PN / 128), 256, 0, stream>>>(
        resbf, wOp, bnTab + 9 * 256, bnTab + 9 * 256 + 128, lifTab, midbf);

    // ---- fc2 + residual -> f32 out [B,N,64] ----
    fc2_final_kernel<<<(int)(PN / 128), 256, 0, stream>>>(
        midbf, wFc2, bnTab + 1 * 256, bnTab + 1 * 256 + 128, feats, (float*)d_out);
}
